// MultiHeadAttention_54992761258378
// MI455X (gfx1250) — matrix operand order repaired and checked by reading, not compiled, not run on hardware
//
#include <hip/hip_runtime.h>

typedef __attribute__((ext_vector_type(16))) _Float16 v16h;
typedef __attribute__((ext_vector_type(8)))  _Float16 v8h;
typedef __attribute__((ext_vector_type(8)))  float    v8f;

#define HH  8
#define DKK 64
#define SS  2048
#define DD  512

__device__ __forceinline__ v8f wmma_f32_f16(v16h a, v16h b, v8f c) {
  // (neg_a, A, neg_b, B, c_mod, C, reuse_a, reuse_b)
  return __builtin_amdgcn_wmma_f32_16x16x32_f16(false, a, false, b, (short)0, c,
                                                false, false);
}

// A operand (16x32 f16, MxK) from row-major LDS tile [rows][ldh].
// Lane L holds row (row0 + L%16); halves: K = {0..7,16..23} (+8 if L>=16), +kbase.
__device__ __forceinline__ v16h frag_a(const _Float16* base, int row0, int ldh,
                                       int kbase) {
  const int lane = threadIdx.x & 31;
  const _Float16* p =
      base + (row0 + (lane & 15)) * ldh + kbase + ((lane & 16) ? 8 : 0);
  v8h lo = *(const v8h*)(p);
  v8h hi = *(const v8h*)(p + 16);
  v16h r;
#pragma unroll
  for (int i = 0; i < 8; ++i) { r[i] = lo[i]; r[i + 8] = hi[i]; }
  return r;
}

// B operand (32x16 f16, KxN) from LDS tile stored [N][ldh(K)] row-major.
// Lane L holds column (col0 + L%16); halves: K = {0..7,16..23} (+8 if L>=16), +kbase.
__device__ __forceinline__ v16h frag_b(const _Float16* base, int col0, int ldh,
                                       int kbase) {
  const int lane = threadIdx.x & 31;
  const _Float16* p =
      base + (col0 + (lane & 15)) * ldh + kbase + ((lane & 16) ? 8 : 0);
  v8h lo = *(const v8h*)(p);
  v8h hi = *(const v8h*)(p + 16);
  v16h r;
#pragma unroll
  for (int i = 0; i < 8; ++i) { r[i] = lo[i]; r[i + 8] = hi[i]; }
  return r;
}

// ---------------------------------------------------------------------------
// GEMM: out = A[M x 512] @ W[512 x 512]^T + bias   (W row-major [N][K])
// mode 0: write f16 head-split [B][H][S][DK];  mode 1: write f32 [M][512]
// ---------------------------------------------------------------------------
#define GM 128
#define GN 64
#define GK 32

__global__ __launch_bounds__(256) void gemm_bias_kernel(
    const float* __restrict__ A, const float* __restrict__ W,
    const float* __restrict__ bias, _Float16* __restrict__ out_h,
    float* __restrict__ out_f, int mode) {
  __shared__ _Float16 As[GM][GK];  // 8 KB
  __shared__ _Float16 Bs[GN][GK];  // 4 KB, stored [N][K]
  const int tid = threadIdx.x, wave = tid >> 5, lane = tid & 31;
  const int m0 = blockIdx.x * GM;
  const int n0 = blockIdx.y * GN;

  v8f acc[4] = {v8f{}, v8f{}, v8f{}, v8f{}};
  for (int k0 = 0; k0 < DD; k0 += GK) {
#pragma unroll 2
    for (int i = tid; i < GM * GK; i += 256) {
      int r = i >> 5, c = i & 31;
      As[r][c] = (_Float16)A[(size_t)(m0 + r) * DD + k0 + c];
    }
#pragma unroll 2
    for (int i = tid; i < GN * GK; i += 256) {
      int r = i >> 5, c = i & 31;
      Bs[r][c] = (_Float16)W[(size_t)(n0 + r) * DD + k0 + c];
    }
    __syncthreads();
    v16h af = frag_a(&As[0][0], wave * 16, GK, 0);
#pragma unroll
    for (int t = 0; t < 4; ++t) {
      v16h bf = frag_b(&Bs[0][0], t * 16, GK, 0);
      acc[t] = wmma_f32_f16(af, bf, acc[t]);
    }
    __syncthreads();
  }

  // Epilogue: C-frag element i -> row = i + (lane<16?0:8), col = lane%16
  const int rbase = m0 + wave * 16 + ((lane & 16) ? 8 : 0);
#pragma unroll
  for (int t = 0; t < 4; ++t) {
    const int n = n0 + t * 16 + (lane & 15);
    const float bv = bias[n];
#pragma unroll
    for (int i = 0; i < 8; ++i) {
      const int m = rbase + i;
      const float val = acc[t][i] + bv;
      if (mode == 0) {
        const int b = m >> 11, s = m & (SS - 1);   // M index = b*S + s
        const int h = n >> 6, dk = n & 63;         // N index = h*DK + dk
        out_h[(((size_t)b * HH + h) * SS + s) * DKK + dk] = (_Float16)val;
      } else {
        out_f[(size_t)m * DD + n] = val;
      }
    }
  }
}

// ---------------------------------------------------------------------------
// Flash attention: per (b,h) and 128-query block; key blocks of 64.
// q/k/v in f16 [B][H][S][DK]; ctx out f32 [B*S][512].
// ---------------------------------------------------------------------------
#define QB 128
#define KBL 64

__global__ __launch_bounds__(256) void attn_kernel(
    const _Float16* __restrict__ qh, const _Float16* __restrict__ kh,
    const _Float16* __restrict__ vh, float* __restrict__ ctx) {
  __shared__ _Float16 Qs[QB][DKK];       // 16 KB  [query][dk]
  __shared__ _Float16 Ks[KBL][DKK];      // 8 KB   [key][dk]  (== B[N][K] for scores)
  __shared__ _Float16 Vt[DKK][KBL];      // 8 KB   [dk][key]  (== B[N][K] for O gemm)
  __shared__ _Float16 Ps[8][16][KBL];    // 16 KB  per-wave P tile [q][key]

  const int tid = threadIdx.x, wave = tid >> 5, lane = tid & 31;
  const int bh = blockIdx.y;             // b*H + h
  const int q0 = blockIdx.x * QB;
  const size_t hb = (size_t)bh * SS * DKK;
  const _Float16* qp = qh + hb + (size_t)q0 * DKK;
  const _Float16* kp = kh + hb;
  const _Float16* vp = vh + hb;

  {  // stage Q block, 16 KB, 16B vector copies
    const float4* src = (const float4*)qp;
    float4* dst = (float4*)&Qs[0][0];
#pragma unroll
    for (int i = tid; i < QB * DKK / 8; i += 256) dst[i] = src[i];
  }

  v8f O[4] = {v8f{}, v8f{}, v8f{}, v8f{}};
  float mrow[8], lrow[8];
#pragma unroll
  for (int i = 0; i < 8; ++i) { mrow[i] = -3.0e38f; lrow[i] = 0.0f; }
  const float scale = 0.125f;  // 1/sqrt(64)

  for (int kb = 0; kb < SS; kb += KBL) {
    __syncthreads();  // previous Ks/Vt consumers done
    {
      const float4* src = (const float4*)(kp + (size_t)kb * DKK);
      float4* dst = (float4*)&Ks[0][0];
      for (int i = tid; i < KBL * DKK / 8; i += 256) dst[i] = src[i];
      const _Float16* vsrc = vp + (size_t)kb * DKK;
      for (int i = tid; i < KBL * DKK; i += 256) {
        int key = i >> 6, dk = i & 63;
        Vt[dk][key] = vsrc[i];  // transpose on the way in
      }
    }
    __syncthreads();

    // scores = Q @ K^T : K-dim = dk (64 -> 2 chunks of 32)
    v8f sc[4] = {v8f{}, v8f{}, v8f{}, v8f{}};
#pragma unroll
    for (int c = 0; c < DKK; c += 32) {
      v16h aq = frag_a(&Qs[0][0], wave * 16, DKK, c);
#pragma unroll
      for (int t = 0; t < 4; ++t) {
        v16h bk = frag_b(&Ks[0][0], t * 16, DKK, c);
        sc[t] = wmma_f32_f16(aq, bk, sc[t]);
      }
    }
#pragma unroll
    for (int t = 0; t < 4; ++t)
#pragma unroll
      for (int i = 0; i < 8; ++i) sc[t][i] *= scale;

    // ---- online softmax (row stats live per 16-lane half, 8 rows/lane-elt)
    float rmax[8];
#pragma unroll
    for (int i = 0; i < 8; ++i)
      rmax[i] = fmaxf(fmaxf(sc[0][i], sc[1][i]), fmaxf(sc[2][i], sc[3][i]));
#pragma unroll
    for (int m = 1; m <= 8; m <<= 1)
#pragma unroll
      for (int i = 0; i < 8; ++i)
        rmax[i] = fmaxf(rmax[i], __shfl_xor(rmax[i], m, 32));

    float rsum[8];
#pragma unroll
    for (int i = 0; i < 8; ++i) {
      const float mnew = fmaxf(mrow[i], rmax[i]);
      const float alpha = __expf(mrow[i] - mnew);
      mrow[i] = mnew;
      lrow[i] *= alpha;
#pragma unroll
      for (int t = 0; t < 4; ++t) O[t][i] *= alpha;
      rsum[i] = 0.0f;
    }
#pragma unroll
    for (int t = 0; t < 4; ++t)
#pragma unroll
      for (int i = 0; i < 8; ++i) {
        const float p = __expf(sc[t][i] - mrow[i]);
        sc[t][i] = p;
        rsum[i] += p;
      }
#pragma unroll
    for (int m = 1; m <= 8; m <<= 1)
#pragma unroll
      for (int i = 0; i < 8; ++i) rsum[i] += __shfl_xor(rsum[i], m, 32);
#pragma unroll
    for (int i = 0; i < 8; ++i) lrow[i] += rsum[i];

    // C-layout -> A-layout via per-wave LDS tile (wave-private; LDS is
    // in-order within a wave, so no cross-wave sync needed)
    {
      const int rb = (lane & 16) ? 8 : 0;
      const int col = lane & 15;
#pragma unroll
      for (int t = 0; t < 4; ++t)
#pragma unroll
        for (int i = 0; i < 8; ++i)
          Ps[wave][rb + i][t * 16 + col] = (_Float16)sc[t][i];
    }
    asm volatile("" ::: "memory");

    // O += P @ V : K-dim = key (64 -> 2 chunks of 32)
#pragma unroll
    for (int c = 0; c < KBL; c += 32) {
      v16h ap = frag_a(&Ps[wave][0][0], 0, KBL, c);
#pragma unroll
      for (int t = 0; t < 4; ++t) {
        v16h bv = frag_b(&Vt[0][0], t * 16, KBL, c);
        O[t] = wmma_f32_f16(ap, bv, O[t]);
      }
    }
  }

  {  // normalize + scatter into ctx [B*S][512]
    const int b = bh >> 3, h = bh & 7;
    const int rbase = q0 + wave * 16 + ((lane & 16) ? 8 : 0);
    const int col = h * DKK + (lane & 15);
#pragma unroll
    for (int t = 0; t < 4; ++t)
#pragma unroll
      for (int i = 0; i < 8; ++i) {
        const int s = rbase + i;
        ctx[((size_t)b * SS + s) * DD + col + t * 16] = O[t][i] / lrow[i];
      }
  }
}

// ---------------------------------------------------------------------------
// out = LayerNorm(proj + residual) * gamma + beta ; one block per row of 512
// ---------------------------------------------------------------------------
__global__ __launch_bounds__(256) void ln_kernel(
    const float* __restrict__ x, const float* __restrict__ resid,
    const float* __restrict__ gamma, const float* __restrict__ beta,
    float* __restrict__ out) {
  __shared__ float red[256];
  const int row = blockIdx.x, tid = threadIdx.x;
  const size_t base = (size_t)row * DD;
  const float v0 = x[base + tid] + resid[base + tid];
  const float v1 = x[base + tid + 256] + resid[base + tid + 256];

  red[tid] = v0 + v1;
  __syncthreads();
  for (int off = 128; off > 0; off >>= 1) {
    if (tid < off) red[tid] += red[tid + off];
    __syncthreads();
  }
  const float mu = red[0] * (1.0f / 512.0f);
  __syncthreads();

  const float d0 = v0 - mu, d1 = v1 - mu;
  red[tid] = d0 * d0 + d1 * d1;
  __syncthreads();
  for (int off = 128; off > 0; off >>= 1) {
    if (tid < off) red[tid] += red[tid + off];
    __syncthreads();
  }
  const float inv = rsqrtf(red[0] * (1.0f / 512.0f) + 1e-5f);

  out[base + tid] = d0 * inv * gamma[tid] + beta[tid];
  out[base + tid + 256] = d1 * inv * gamma[tid + 256] + beta[tid + 256];
}

// ---------------------------------------------------------------------------
extern "C" void kernel_launch(void* const* d_in, const int* in_sizes, int n_in,
                              void* d_out, int out_size, void* d_ws,
                              size_t ws_size, hipStream_t stream) {
  (void)in_sizes; (void)n_in; (void)out_size; (void)ws_size;
  const float* Q  = (const float*)d_in[0];
  const float* K  = (const float*)d_in[1];
  const float* V  = (const float*)d_in[2];
  const float* Wq = (const float*)d_in[3];
  const float* bq = (const float*)d_in[4];
  const float* Wk = (const float*)d_in[5];
  const float* bk = (const float*)d_in[6];
  const float* Wv = (const float*)d_in[7];
  const float* bv = (const float*)d_in[8];
  const float* Wo = (const float*)d_in[9];
  const float* bo = (const float*)d_in[10];
  const float* gamma = (const float*)d_in[11];
  const float* beta  = (const float*)d_in[12];
  float* out = (float*)d_out;

  // Workspace layout (bytes):
  //   [0,16M)  qh f16   [16M,32M) kh f16   [32M,48M) vh f16
  //   [48M,80M) ctx f32
  //   proj f32 reuses [0,32M) (qh/kh dead by then)
  char* ws = (char*)d_ws;
  _Float16* qh = (_Float16*)(ws);
  _Float16* kh = (_Float16*)(ws + (size_t)16 * 1024 * 1024);
  _Float16* vh = (_Float16*)(ws + (size_t)32 * 1024 * 1024);
  float* ctx   = (float*)(ws + (size_t)48 * 1024 * 1024);
  float* proj  = (float*)(ws);

  const int M = 8 * SS;  // 16384 rows
  dim3 ggrid(M / GM, DD / GN);
  gemm_bias_kernel<<<ggrid, 256, 0, stream>>>(Q, Wq, bq, qh, nullptr, 0);
  gemm_bias_kernel<<<ggrid, 256, 0, stream>>>(K, Wk, bk, kh, nullptr, 0);
  gemm_bias_kernel<<<ggrid, 256, 0, stream>>>(V, Wv, bv, vh, nullptr, 0);

  dim3 agrid(SS / QB, 8 * HH);  // (16 query blocks, 64 b*h)
  attn_kernel<<<agrid, 256, 0, stream>>>(qh, kh, vh, ctx);

  gemm_bias_kernel<<<ggrid, 256, 0, stream>>>(ctx, Wo, bo, nullptr, proj, 1);

  ln_kernel<<<M, 256, 0, stream>>>(proj, Q, gamma, beta, out);
}
